// TernaryLinear_28312424415895
// MI455X (gfx1250) — matrix operand order repaired and checked by reading, not compiled, not run on hardware
//
#include <hip/hip_runtime.h>

typedef __attribute__((ext_vector_type(16))) _Float16 v16h;
typedef __attribute__((ext_vector_type(8)))  _Float16 v8h;
typedef __attribute__((ext_vector_type(8)))  float    v8f;

#define D_IN   512
#define D_OUT  512
#define K_STEPS (D_IN / 32)          // 16 wmma K-steps of 32
#define LDS_STRIDE 520               // 512 + 8 halves pad -> conflict-free b128 reads
#define THRESHOLD 0.05f

// ---------------------------------------------------------------------------
// Kernel 1: ternarize f32 weights and pack directly into WMMA B-fragment
// order:  packed[ntile][kstep][lane][e] , 16 halves (32B) per lane.
//   B 32x16 fragment mapping used here (contiguous K order, same as A):
//     lane<16 : elem e = Wt[ntile*16 + lane][kstep*32 + e]
//     lane>=16: elem e = Wt[ntile*16 + lane-16][kstep*32 + 16 + e]
// GEMM-side load becomes base + lane*32B -> one contiguous 1KB burst per wave.
// Total 512KB, L2-resident for the whole GEMM.
// ---------------------------------------------------------------------------
__global__ void pack_b_frags_kernel(const float* __restrict__ w,
                                    _Float16* __restrict__ packed, int n) {
    int idx = blockIdx.x * blockDim.x + threadIdx.x;
    if (idx >= n) return;
    int e     = idx & 15;            // element within lane
    int lane  = (idx >> 4) & 31;
    int kstep = (idx >> 9) & 15;
    int ntile = idx >> 13;
    int row = ntile * 16 + (lane & 15);          // output-feature index
    int k   = kstep * 32 + (lane >> 4) * 16 + e; // input-feature index
    float v = w[row * D_IN + k];
    float t = (fabsf(v) < THRESHOLD) ? 0.0f : (v > 0.0f ? 1.0f : -1.0f);
    packed[idx] = (_Float16)t;
}

// ---------------------------------------------------------------------------
// Kernel 2: WMMA GEMM. Block = 256 threads = 8 waves.
// Block tile: M=32 (x slab in LDS, f32->f16), N=512.
// Wave w: N in [64w, 64w+64) as 4 subtiles x 2 M-subtiles -> acc[2][4].
// Each B fragment feeds TWO wmmas; 8 independent accumulation chains per
// k-step hide L2/LDS latency. K-loop unroll bounded to 4 steps (32 wmma per
// body) so the scheduler's prefetch window stays register-frugal without
// spilling (launch-bounds clamping was shown to cause scratch spills).
// ---------------------------------------------------------------------------
__global__ void __launch_bounds__(256)
ternary_linear_wmma_kernel(const float* __restrict__ x,
                           const _Float16* __restrict__ bfrag,
                           const float* __restrict__ bias,
                           float* __restrict__ out) {
    __shared__ _Float16 As[32 * LDS_STRIDE];   // 32 rows x (512+pad) f16 = 33.3 KB

    const int tid  = threadIdx.x;
    const int lane = tid & 31;
    const int wave = tid >> 5;
    const int half = lane >> 4;     // 0: lanes 0-15, 1: lanes 16-31
    const int l15  = lane & 15;
    const long m0  = (long)blockIdx.x * 32;

    // ---- Stage x[m0..m0+31][0..511] into LDS as f16.
    // 4096 float4 loads across 256 threads (16 each), 128B-coalesced.
    #pragma unroll
    for (int j = 0; j < 16; ++j) {
        int idx = tid + j * 256;            // 0..4095
        int row = idx >> 7;                 // 128 float4 per row
        int c4  = idx & 127;
        const float4 v = ((const float4*)(x + (m0 + row) * D_IN))[c4];
        _Float16* dst = &As[row * LDS_STRIDE + c4 * 4];
        dst[0] = (_Float16)v.x;  dst[1] = (_Float16)v.y;
        dst[2] = (_Float16)v.z;  dst[3] = (_Float16)v.w;
    }
    __syncthreads();

    v8f acc[2][4];
    #pragma unroll
    for (int ms = 0; ms < 2; ++ms)
        #pragma unroll
        for (int ns = 0; ns < 4; ++ns) acc[ms][ns] = (v8f)0.0f;

    // A-fragment rows (contiguous K order, matching packed B):
    //   elems 0..15 = K[kbase + 16*half + 0..15]
    const _Float16* arow0 = &As[l15 * LDS_STRIDE];          // M rows 0..15
    const _Float16* arow1 = &As[(l15 + 16) * LDS_STRIDE];   // M rows 16..31
    const int ntile0 = wave * 4;

    #pragma unroll 4
    for (int kb = 0; kb < K_STEPS; ++kb) {
        const int kbase = kb * 32;

        v8h a0lo = *(const v8h*)(arow0 + kbase + half * 16);
        v8h a0hi = *(const v8h*)(arow0 + kbase + half * 16 + 8);
        v16h a0  = __builtin_shufflevector(a0lo, a0hi,
                      0,1,2,3,4,5,6,7,8,9,10,11,12,13,14,15);
        v8h a1lo = *(const v8h*)(arow1 + kbase + half * 16);
        v8h a1hi = *(const v8h*)(arow1 + kbase + half * 16 + 8);
        v16h a1  = __builtin_shufflevector(a1lo, a1hi,
                      0,1,2,3,4,5,6,7,8,9,10,11,12,13,14,15);

        #pragma unroll
        for (int ns = 0; ns < 4; ++ns) {
            // Packed fragment: contiguous 1KB per (ntile,kstep), lane*32B.
            const _Float16* bp = bfrag
                + (((size_t)(ntile0 + ns) * K_STEPS + kb) << 9) + lane * 16;
            v8h b0 = *(const v8h*)bp;
            v8h b1 = *(const v8h*)(bp + 8);
            v16h b = __builtin_shufflevector(b0, b1,
                        0,1,2,3,4,5,6,7,8,9,10,11,12,13,14,15);

            acc[0][ns] = __builtin_amdgcn_wmma_f32_16x16x32_f16(
                false, a0, false, b, (short)0, acc[0][ns], false, false);
            acc[1][ns] = __builtin_amdgcn_wmma_f32_16x16x32_f16(
                false, a1, false, b, (short)0, acc[1][ns], false, false);
        }
    }

    // ---- Store: C/D f32 layout -> VGPR i holds (M = i + 8*half, N = l15).
    #pragma unroll
    for (int ms = 0; ms < 2; ++ms) {
        #pragma unroll
        for (int ns = 0; ns < 4; ++ns) {
            const int n = wave * 64 + ns * 16 + l15;
            const float bv = bias[n];
            long base = (m0 + ms * 16 + half * 8) * (long)D_OUT + n;
            #pragma unroll
            for (int i = 0; i < 8; ++i)
                out[base + (long)i * D_OUT] = acc[ms][ns][i] + bv;
        }
    }
}

extern "C" void kernel_launch(void* const* d_in, const int* in_sizes, int n_in,
                              void* d_out, int out_size, void* d_ws, size_t ws_size,
                              hipStream_t stream) {
    const float* x    = (const float*)d_in[0];
    const float* w    = (const float*)d_in[1];
    const float* bias = (const float*)d_in[2];
    float*       out  = (float*)d_out;
    _Float16*    bfrag = (_Float16*)d_ws;         // 512 KB packed B fragments

    const int wn = in_sizes[1];                   // 262144 weight elements
    pack_b_frags_kernel<<<(wn + 255) / 256, 256, 0, stream>>>(w, bfrag, wn);

    const int M = in_sizes[0] / D_IN;             // 65536 rows
    ternary_linear_wmma_kernel<<<dim3(M / 32), 256, 0, stream>>>(x, bfrag, bias, out);
}
